// lstm_crf_89773406421379
// MI455X (gfx1250) — compile-verified, matrix operand order checked by reading, not run on hardware
//
#include <hip/hip_runtime.h>
#include <hip/hip_bf16.h>

typedef __attribute__((ext_vector_type(16))) __bf16 v16bf;
typedef __attribute__((ext_vector_type(8)))  __bf16 v8bf;
typedef __attribute__((ext_vector_type(8)))  float  v8f;
typedef __attribute__((ext_vector_type(4)))  float  v4f;
typedef __attribute__((ext_vector_type(4)))  unsigned int v4u;
typedef __attribute__((ext_vector_type(8)))  int v8i;
typedef __attribute__((ext_vector_type(4)))  int v4i;

#if defined(__has_builtin)
#if __has_builtin(__builtin_amdgcn_tensor_load_to_lds) && __has_builtin(__builtin_amdgcn_s_wait_tensorcnt)
#define HAVE_TDM 1
#endif
#endif
#ifndef HAVE_TDM
#define HAVE_TDM 0
#endif

namespace {

constexpr int Bc  = 64;    // batch
constexpr int Tc  = 512;   // time
constexpr int Ec  = 128;   // embed dim
constexpr int HPc = 128;   // hidden per direction
constexpr int NG  = 512;   // 4*HP gate width
constexpr int KK  = 32;    // tags
constexpr int SOS = 2, EOS = 3;

// Fragment k mapping per ISA 7.12.2 (16-bit A/B): lanes 0-15 hold K {0..7,16..23},
// lanes 16-31 hold K {8..15,24..31}.  Each half-fragment is 8 contiguous bf16 = 16B.

// Load v16bf fragment from bf16 memory (global or LDS), base = row start + kk.
__device__ __forceinline__ v16bf load_frag_bf16(const __bf16* base, int lane) {
  int kbase = (lane >> 4) << 3;                 // 0 or 8
  v8bf lo = *(const v8bf*)(base + kbase);       // k: kbase .. kbase+7
  v8bf hi = *(const v8bf*)(base + kbase + 16);  // k: kbase+16 .. kbase+23
  v16bf r;
#pragma unroll
  for (int e = 0; e < 8; ++e) { r[e] = lo[e]; r[e + 8] = hi[e]; }
  return r;
}

// Load v16bf fragment from f32 memory with on-the-fly bf16 convert.
__device__ __forceinline__ v16bf load_frag_f32(const float* base, int lane) {
  int kbase = (lane >> 4) << 3;
  v4f a = *(const v4f*)(base + kbase);
  v4f b = *(const v4f*)(base + kbase + 4);
  v4f c = *(const v4f*)(base + kbase + 16);
  v4f d = *(const v4f*)(base + kbase + 20);
  v16bf r;
#pragma unroll
  for (int e = 0; e < 4; ++e) {
    r[e]      = (__bf16)a[e];
    r[e + 4]  = (__bf16)b[e];
    r[e + 8]  = (__bf16)c[e];
    r[e + 12] = (__bf16)d[e];
  }
  return r;
}

__device__ __forceinline__ float sigmf(float x) { return 1.f / (1.f + __expf(-x)); }

// ---------------------------------------------------------------- lengths
__global__ void lengths_kernel(const int* __restrict__ x, int* __restrict__ lengths) {
  __shared__ int red[256];
  int b = blockIdx.x, cnt = 0;
  for (int t = threadIdx.x; t < Tc; t += 256) cnt += (x[b * Tc + t] > 0) ? 1 : 0;
  red[threadIdx.x] = cnt;
  __syncthreads();
  for (int s = 128; s > 0; s >>= 1) {
    if (threadIdx.x < s) red[threadIdx.x] += red[threadIdx.x + s];
    __syncthreads();
  }
  if (threadIdx.x == 0) lengths[b] = red[0];
}

// ---------------------------------------------------------------- f32 -> bf16 convert
__global__ void cvt_bf16_kernel(const float* __restrict__ in, __bf16* __restrict__ out, int n) {
  int i = blockIdx.x * 256 + threadIdx.x;
  if (i < n) out[i] = (__bf16)in[i];
}

// ---------------------------------------------------------------- embedding gather -> bf16
__global__ void gather_kernel(const int* __restrict__ x, const float* __restrict__ embed,
                              __bf16* __restrict__ out) {
  int i = blockIdx.x * 256 + threadIdx.x;     // grid exactly covers B*T*E
  int ti = i >> 7;                            // / E (=128)
  int e  = i & 127;
  out[i] = (__bf16)embed[(size_t)x[ti] * Ec + e];
}

// ---------------------------------------------------------------- Out(bf16) = A(MxKd bf16) * W(NxKd f32)^T + bias
__global__ __launch_bounds__(256)
void xproj_gemm(const __bf16* __restrict__ A, int lda,
                const float* __restrict__ W, const float* __restrict__ bias,
                __bf16* __restrict__ Out, int M, int N, int Kd) {
  int lane = threadIdx.x & 31;
  int wave = blockIdx.x * (blockDim.x >> 5) + (threadIdx.x >> 5);
  int ntiles = N >> 4;
  int total  = (M >> 4) * ntiles;
  if (wave >= total) return;
  int nt = wave % ntiles, mt = wave / ntiles;
  int n = nt * 16 + (lane & 15);
  int m = mt * 16 + (lane & 15);
  v8f acc;
  float bv = bias[n];
#pragma unroll
  for (int r = 0; r < 8; ++r) acc[r] = bv;
  for (int kk = 0; kk < Kd; kk += 32) {
    v16bf a   = load_frag_bf16(A + (size_t)m * lda + kk, lane);
    v16bf bfr = load_frag_f32(W + (size_t)n * Kd + kk, lane);
    acc = __builtin_amdgcn_wmma_f32_16x16x32_bf16(false, a, false, bfr, (short)0, acc,
                                                  false, false);
  }
#pragma unroll
  for (int r = 0; r < 8; ++r) {
    int row = mt * 16 + r + ((lane >> 4) << 3);
    Out[(size_t)row * N + n] = (__bf16)acc[r];
  }
}

// ---------------------------------------------------------------- emissions (f32), masked by length
__global__ __launch_bounds__(256)
void outproj_gemm(const __bf16* __restrict__ A, int lda,
                  const float* __restrict__ W, const float* __restrict__ bias,
                  float* __restrict__ Out, const int* __restrict__ lengths,
                  int M, int N, int Kd) {
  int lane = threadIdx.x & 31;
  int wave = blockIdx.x * (blockDim.x >> 5) + (threadIdx.x >> 5);
  int ntiles = N >> 4;
  int total  = (M >> 4) * ntiles;
  if (wave >= total) return;
  int nt = wave % ntiles, mt = wave / ntiles;
  int n = nt * 16 + (lane & 15);
  int m = mt * 16 + (lane & 15);
  v8f acc;
  float bv = bias[n];
#pragma unroll
  for (int r = 0; r < 8; ++r) acc[r] = bv;
  for (int kk = 0; kk < Kd; kk += 32) {
    v16bf a   = load_frag_bf16(A + (size_t)m * lda + kk, lane);
    v16bf bfr = load_frag_f32(W + (size_t)n * Kd + kk, lane);
    acc = __builtin_amdgcn_wmma_f32_16x16x32_bf16(false, a, false, bfr, (short)0, acc,
                                                  false, false);
  }
#pragma unroll
  for (int r = 0; r < 8; ++r) {
    int row = mt * 16 + r + ((lane >> 4) << 3);
    int b = row >> 9, t = row & 511;
    float v = (t < lengths[b]) ? acc[r] : 0.f;
    Out[(size_t)row * N + n] = v;
  }
}

// ---------------------------------------------------------------- LSTM recurrence, one block per direction
// Dynamic LDS: Wh bf16 (512x128 = 128KB, at offset 0 so the TDM D# lds_addr=0) +
//              h state bf16 (64x128 = 16KB) + lengths.
__global__ __launch_bounds__(512)
void lstm_rec_kernel(const __bf16* __restrict__ whf, const __bf16* __restrict__ whb,
                     const __bf16* __restrict__ xpf, const __bf16* __restrict__ xpb,
                     __bf16* __restrict__ hout, const int* __restrict__ lengths) {
  extern __shared__ unsigned char smem[];
  __bf16* wh_lds = (__bf16*)smem;                                        // NG*HPc
  __bf16* h_lds  = (__bf16*)(smem + (size_t)NG * HPc * 2);               // Bc*HPc
  int*    len_s  = (int*)(smem + (size_t)NG * HPc * 2 + (size_t)Bc * HPc * 2);

  int dir = blockIdx.x;                         // 0 = forward, 1 = backward
  const __bf16* wh = dir ? whb : whf;
  const __bf16* xp = dir ? xpb : xpf;
  int hoff = dir ? HPc : 0;

  int tid = threadIdx.x;
#if HAVE_TDM
  // DMA the 512x128 bf16 Wh tile into LDS via the Tensor Data Mover.
  if (tid < 32) {
    unsigned long long ga = (unsigned long long)wh;
    v4u g0;
    g0[0] = 1u;                                            // count=1, user descriptor
    g0[1] = 0u;                                            // lds_addr = 0
    g0[2] = (unsigned)(ga & 0xFFFFFFFFu);                  // global_addr lo
    g0[3] = (unsigned)((ga >> 32) & 0x1FFFFFFu) | (2u << 30);  // addr hi | type=2
    v8i g1;
    g1[0] = 0x00010000;                                    // data_size = 1 (2 bytes)
    g1[1] = (int)(128u << 16);                             // tensor_dim0 = 128
    g1[2] = (int)(512u << 16);                             // tensor_dim1 = 512
    g1[3] = (int)(128u << 16);                             // tile_dim0 = 128
    g1[4] = 512;                                           // tile_dim1 = 512
    g1[5] = 128;                                           // tensor_dim0_stride = 128
    g1[6] = 0;
    g1[7] = 0;
    v4i z4 = {0, 0, 0, 0};
#if __clang_major__ >= 23
    v8i z8 = {0, 0, 0, 0, 0, 0, 0, 0};
    __builtin_amdgcn_tensor_load_to_lds(g0, g1, z4, z4, z8, 0);
#else
    __builtin_amdgcn_tensor_load_to_lds(g0, g1, z4, z4, 0);
#endif
    __builtin_amdgcn_s_wait_tensorcnt(0);
  }
#else
  for (int i = tid * 8; i < NG * HPc; i += 512 * 8)
    *(v8bf*)(wh_lds + i) = *(const v8bf*)(wh + i);
#endif
  for (int i = tid; i < Bc * HPc; i += 512) h_lds[i] = (__bf16)0.f;
  for (int i = tid; i < Bc; i += 512) len_s[i] = lengths[i];
  __syncthreads();

  int lane = tid & 31, wave = tid >> 5;
  float c[2][8];
#pragma unroll
  for (int u = 0; u < 2; ++u)
#pragma unroll
    for (int r = 0; r < 8; ++r) c[u][r] = 0.f;

  for (int t = 0; t < Tc; ++t) {
    v8f acc[2][4];
    for (int u = 0; u < 2; ++u) {
      int unit = wave * 2 + u;
      int mi = unit >> 3, j = unit & 7;        // mi: batch tile 0..3, j: col tile 0..7
      // init accumulators with precomputed x@Wi^T + b
#pragma unroll
      for (int g = 0; g < 4; ++g) {
        int n = (g * 8 + j) * 16 + (lane & 15);
#pragma unroll
        for (int r = 0; r < 8; ++r) {
          int b = mi * 16 + r + ((lane >> 4) << 3);
          int len = len_s[b];
          int tt = dir ? ((t < len) ? (len - 1 - t) : t) : t;
          acc[u][g][r] = (float)xp[((size_t)b * Tc + tt) * NG + n];
        }
      }
      // prefetch next timestep's projection slice
      if (t + 1 < Tc) {
        int b0 = mi * 16 + (lane & 15);
        int len0 = len_s[b0];
        int t2 = dir ? ((t + 1 < len0) ? (len0 - 2 - t) : (t + 1)) : (t + 1);
        __builtin_prefetch(&xp[((size_t)b0 * Tc + t2) * NG + j * 16], 0, 1);
      }
      // h_prev @ Wh^T
      for (int kk = 0; kk < HPc; kk += 32) {
        int m = mi * 16 + (lane & 15);
        v16bf a = load_frag_bf16(h_lds + m * HPc + kk, lane);
#pragma unroll
        for (int g = 0; g < 4; ++g) {
          int n = (g * 8 + j) * 16 + (lane & 15);
          v16bf bfr = load_frag_bf16(wh_lds + n * HPc + kk, lane);
          acc[u][g] = __builtin_amdgcn_wmma_f32_16x16x32_bf16(false, a, false, bfr, (short)0,
                                                              acc[u][g], false, false);
        }
      }
    }
    __syncthreads();   // all waves done reading previous h

    for (int u = 0; u < 2; ++u) {
      int unit = wave * 2 + u;
      int mi = unit >> 3, j = unit & 7;
      int ncol = j * 16 + (lane & 15);
#pragma unroll
      for (int r = 0; r < 8; ++r) {
        int b = mi * 16 + r + ((lane >> 4) << 3);
        float zi = acc[u][0][r], zf = acc[u][1][r], zg = acc[u][2][r], zo = acc[u][3][r];
        float cn = sigmf(zf) * c[u][r] + sigmf(zi) * tanhf(zg);
        c[u][r] = cn;
        float hv = sigmf(zo) * tanhf(cn);
        __bf16 hb = (__bf16)hv;
        h_lds[b * HPc + ncol] = hb;
        int len = len_s[b];
        int tt = dir ? ((t < len) ? (len - 1 - t) : t) : t;
        hout[((size_t)b * Tc + tt) * (2 * HPc) + hoff + ncol] = hb;
      }
    }
    __syncthreads();   // new h visible for next step
  }
}

// ---------------------------------------------------------------- CRF forward + gold path
__global__ __launch_bounds__(256)
void crf_kernel(const float* __restrict__ scores, const int* __restrict__ y,
                const float* __restrict__ trans, const int* __restrict__ lengths,
                float* __restrict__ out) {
  __shared__ float sc[2][Bc][KK];
  __shared__ float tr[KK * KK];
  __shared__ int len_s[Bc];
  int tid = threadIdx.x;
  for (int i = tid; i < KK * KK; i += 256) tr[i] = trans[i];
  for (int i = tid; i < Bc; i += 256) len_s[i] = lengths[i];
  for (int i = tid; i < Bc * KK; i += 256) {
    int b = i >> 5, k = i & 31;
    sc[0][b][k] = (k == SOS) ? 0.f : -10000.f;
  }
  __syncthreads();
  int cur = 0;
  for (int t = 0; t < Tc; ++t) {
    int nxt = cur ^ 1;
    for (int i = tid; i < Bc * KK; i += 256) {
      int b = i >> 5, k = i & 31;
      float v = sc[cur][b][k];
      if (t < len_s[b]) {
        float mx = -1e30f;
#pragma unroll
        for (int j = 0; j < KK; ++j) mx = fmaxf(mx, sc[cur][b][j] + tr[k * KK + j]);
        float s = 0.f;
#pragma unroll
        for (int j = 0; j < KK; ++j) s += __expf(sc[cur][b][j] + tr[k * KK + j] - mx);
        v = mx + __logf(s) + scores[((size_t)b * Tc + t) * KK + k];
      }
      sc[nxt][b][k] = v;
    }
    __syncthreads();
    cur = nxt;
  }
  if (tid < Bc) {
    int b = tid;
    float mx = -1e30f;
#pragma unroll
    for (int k = 0; k < KK; ++k) mx = fmaxf(mx, sc[cur][b][k] + tr[EOS * KK + k]);
    float s = 0.f;
#pragma unroll
    for (int k = 0; k < KK; ++k) s += __expf(sc[cur][b][k] + tr[EOS * KK + k] - mx);
    float logZ = mx + __logf(s);
    int len = len_s[b];
    float gold = 0.f;
    for (int t = 0; t < len; ++t) {
      int yt1 = y[b * (Tc + 1) + t + 1];
      int yt  = y[b * (Tc + 1) + t];
      gold += scores[((size_t)b * Tc + t) * KK + yt1] + tr[yt1 * KK + yt];
    }
    int last = y[b * (Tc + 1) + len];
    gold += tr[EOS * KK + last];
    out[b] = logZ - gold;
  }
}

}  // namespace

extern "C" void kernel_launch(void* const* d_in, const int* in_sizes, int n_in,
                              void* d_out, int out_size, void* d_ws, size_t ws_size,
                              hipStream_t stream) {
  (void)in_sizes; (void)n_in; (void)out_size; (void)ws_size;
  const int*   x      = (const int*)d_in[0];
  const int*   y      = (const int*)d_in[1];
  const float* embed  = (const float*)d_in[2];
  const float* wi_l0f = (const float*)d_in[3];
  const float* wh_l0f = (const float*)d_in[4];
  const float* b_l0f  = (const float*)d_in[5];
  const float* wi_l0b = (const float*)d_in[6];
  const float* wh_l0b = (const float*)d_in[7];
  const float* b_l0b  = (const float*)d_in[8];
  const float* wi_l1f = (const float*)d_in[9];
  const float* wh_l1f = (const float*)d_in[10];
  const float* b_l1f  = (const float*)d_in[11];
  const float* wi_l1b = (const float*)d_in[12];
  const float* wh_l1b = (const float*)d_in[13];
  const float* b_l1b  = (const float*)d_in[14];
  const float* w_out  = (const float*)d_in[15];
  const float* b_out  = (const float*)d_in[16];
  const float* trans  = (const float*)d_in[17];

  unsigned char* ws = (unsigned char*)d_ws;
  size_t off = 0;
  auto take = [&](size_t bytes) -> void* {
    void* p = ws + off;
    off = (off + bytes + 255) & ~(size_t)255;
    return p;
  };
  int*    lengths = (int*)take((size_t)Bc * 4);
  __bf16* emb     = (__bf16*)take((size_t)Bc * Tc * Ec * 2);
  __bf16* h0      = (__bf16*)take((size_t)Bc * Tc * 2 * HPc * 2);
  __bf16* h1      = (__bf16*)take((size_t)Bc * Tc * 2 * HPc * 2);
  __bf16* xpf     = (__bf16*)take((size_t)Bc * Tc * NG * 2);
  __bf16* xpb     = (__bf16*)take((size_t)Bc * Tc * NG * 2);
  float*  scores  = (float*)take((size_t)Bc * Tc * KK * 4);
  __bf16* wh0f_bf = (__bf16*)take((size_t)NG * HPc * 2);
  __bf16* wh0b_bf = (__bf16*)take((size_t)NG * HPc * 2);
  __bf16* wh1f_bf = (__bf16*)take((size_t)NG * HPc * 2);
  __bf16* wh1b_bf = (__bf16*)take((size_t)NG * HPc * 2);

  lengths_kernel<<<Bc, 256, 0, stream>>>(x, lengths);
  gather_kernel<<<(Bc * Tc * Ec) / 256, 256, 0, stream>>>(x, embed, emb);

  // Wh f32 -> bf16 (so the TDM can DMA them straight into LDS)
  {
    int n = NG * HPc, blocks = (n + 255) / 256;
    cvt_bf16_kernel<<<blocks, 256, 0, stream>>>(wh_l0f, wh0f_bf, n);
    cvt_bf16_kernel<<<blocks, 256, 0, stream>>>(wh_l0b, wh0b_bf, n);
    cvt_bf16_kernel<<<blocks, 256, 0, stream>>>(wh_l1f, wh1f_bf, n);
    cvt_bf16_kernel<<<blocks, 256, 0, stream>>>(wh_l1b, wh1b_bf, n);
  }

  const int Mrows = Bc * Tc;
  {  // layer-0 input projections (Kd = E = 128)
    int tiles = (Mrows / 16) * (NG / 16);
    int blocks = (tiles + 7) / 8;
    xproj_gemm<<<blocks, 256, 0, stream>>>(emb, Ec, wi_l0f, b_l0f, xpf, Mrows, NG, Ec);
    xproj_gemm<<<blocks, 256, 0, stream>>>(emb, Ec, wi_l0b, b_l0b, xpb, Mrows, NG, Ec);
  }

  constexpr size_t DYN = (size_t)NG * HPc * 2 + (size_t)Bc * HPc * 2 + (size_t)Bc * 4;  // 144.25 KB
  hipFuncSetAttribute((const void*)lstm_rec_kernel,
                      hipFuncAttributeMaxDynamicSharedMemorySize, (int)DYN);

  lstm_rec_kernel<<<2, 512, DYN, stream>>>(wh0f_bf, wh0b_bf, xpf, xpb, h0, lengths);

  {  // layer-1 input projections (Kd = 2*HP = 256)
    int tiles = (Mrows / 16) * (NG / 16);
    int blocks = (tiles + 7) / 8;
    xproj_gemm<<<blocks, 256, 0, stream>>>(h0, 2 * HPc, wi_l1f, b_l1f, xpf, Mrows, NG, 2 * HPc);
    xproj_gemm<<<blocks, 256, 0, stream>>>(h0, 2 * HPc, wi_l1b, b_l1b, xpb, Mrows, NG, 2 * HPc);
  }

  lstm_rec_kernel<<<2, 512, DYN, stream>>>(wh1f_bf, wh1b_bf, xpf, xpb, h1, lengths);

  {  // emission projection N = K = 32, masked
    int tiles = (Mrows / 16) * (KK / 16);
    int blocks = (tiles + 7) / 8;
    outproj_gemm<<<blocks, 256, 0, stream>>>(h1, 2 * HPc, w_out, b_out, scores, lengths,
                                             Mrows, KK, 2 * HPc);
  }

  crf_kernel<<<1, 256, 0, stream>>>(scores, y, trans, lengths, (float*)d_out);
}
